// GatedAttentionPool_15290083574044
// MI455X (gfx1250) — compile-verified, matrix operand order checked by reading, not run on hardware
//
#include <hip/hip_runtime.h>
#include <math.h>

typedef __attribute__((ext_vector_type(16))) __bf16 v16bf;
typedef __attribute__((ext_vector_type(8)))  __bf16 v8bf;
typedef __attribute__((ext_vector_type(8)))  float  v8f;
typedef __attribute__((ext_vector_type(4)))  float  v4f;

#define N_INST 65536
#define DDIM   1024
#define ADIM   256
#define NBAG   64
#define ROWS_PER_WG 64   // 4 row-blocks of 16 rows
#define KSTEP  32

// ---------------------------------------------------------------------------
// Kernel 1: transpose + convert weights to bf16:  Wt[a][k] = (bf16) W[k][a]
// Only 2 MB of traffic; negligible vs the GEMM phase.
// ---------------------------------------------------------------------------
__global__ void wcvt_kernel(const float* __restrict__ Vw, const float* __restrict__ Uw,
                            __bf16* __restrict__ Vt, __bf16* __restrict__ Ut) {
    int idx = blockIdx.x * blockDim.x + threadIdx.x;
    if (idx >= ADIM * DDIM) return;
    int a = idx >> 10;          // / DDIM
    int k = idx & (DDIM - 1);   // % DDIM
    Vt[idx] = (__bf16)Vw[k * ADIM + a];
    Ut[idx] = (__bf16)Uw[k * ADIM + a];
}

// ---------------------------------------------------------------------------
// Kernel 2: per-instance attention logits via bf16 WMMA.
//   wave layout: 8 waves = 4 row-blocks x {V-side, U-side}
//   each wave: C = H_tile(16xK) @ W(Kx256) accumulated f32, K in steps of 32
// ---------------------------------------------------------------------------
__global__ __launch_bounds__(256, 1)
void logits_kernel(const float* __restrict__ H,
                   const __bf16* __restrict__ Vt, const __bf16* __restrict__ Ut,
                   const float* __restrict__ Vb, const float* __restrict__ Ub,
                   const float* __restrict__ ww, float* __restrict__ s_out) {
    __shared__ float ldsX[ROWS_PER_WG * ADIM];   // 64 KB: tanh(X+Vb)*ww staging

    const int tid  = threadIdx.x;
    const int wave = tid >> 5;          // 0..7
    const int lane = tid & 31;
    const int vsel = wave & 1;          // 0 => V/tanh side, 1 => U/sigmoid side
    const int r    = wave >> 1;         // row block 0..3
    const int m    = lane & 15;         // row (A) / column (B) within tile
    const int hi   = lane >> 4;         // K-interleave half

    const int row0 = blockIdx.x * ROWS_PER_WG + r * 16;
    const __bf16* __restrict__ Wt = vsel ? Ut : Vt;

    v8f acc[16] = {};                   // 16 x (16x16 f32) = 16 rows x 256 cols

    // A-fragment base: lane holds row m, K in {0..7,16..23} (+8 for hi lanes)
    const float* hrow = H + (size_t)(row0 + m) * DDIM + hi * 8;

    for (int kb = 0; kb < DDIM; kb += KSTEP) {
        const float* hp = hrow + kb;
        v4f a0 = *(const v4f*)(hp);
        v4f a1 = *(const v4f*)(hp + 4);
        v4f a2 = *(const v4f*)(hp + 16);
        v4f a3 = *(const v4f*)(hp + 20);
        v16bf afrag;
        #pragma unroll
        for (int i = 0; i < 4; ++i) {
            afrag[i]      = (__bf16)a0[i];
            afrag[4 + i]  = (__bf16)a1[i];
            afrag[8 + i]  = (__bf16)a2[i];
            afrag[12 + i] = (__bf16)a3[i];
        }
        #pragma unroll
        for (int t = 0; t < 16; ++t) {
            // B-fragment: lane holds column t*16+m, same K interleave; the
            // bf16 weights are transposed so each half is one b128 load.
            const __bf16* wp = Wt + (size_t)(t * 16 + m) * DDIM + kb + hi * 8;
            v8bf blo = *(const v8bf*)(wp);
            v8bf bhi = *(const v8bf*)(wp + 16);
            v16bf bfrag = __builtin_shufflevector(blo, bhi,
                0,1,2,3,4,5,6,7,8,9,10,11,12,13,14,15);
            acc[t] = __builtin_amdgcn_wmma_f32_16x16x32_bf16(
                false, afrag, false, bfrag, (short)0, acc[t], false, false);
        }
    }

    // Epilogue: gate = tanh(X+Vb) * sigmoid(Y+Ub); s = gate . ww
    // C layout: element e of acc[t] is row (hi*8+e), col t*16+m.
    if (vsel == 0) {
        #pragma unroll
        for (int t = 0; t < 16; ++t) {
            const int col  = t * 16 + m;
            const float bi = Vb[col];
            const float w  = ww[col];
            #pragma unroll
            for (int e = 0; e < 8; ++e) {
                const int rr = r * 16 + hi * 8 + e;
                ldsX[rr * ADIM + col] = tanhf(acc[t][e] + bi) * w;
            }
        }
    }
    __syncthreads();
    if (vsel == 1) {
        float part[8] = {0.f, 0.f, 0.f, 0.f, 0.f, 0.f, 0.f, 0.f};
        #pragma unroll
        for (int t = 0; t < 16; ++t) {
            const int col  = t * 16 + m;
            const float bi = Ub[col];
            #pragma unroll
            for (int e = 0; e < 8; ++e) {
                const int rr = r * 16 + hi * 8 + e;
                const float sg = 1.0f / (1.0f + expf(-(acc[t][e] + bi)));
                part[e] += ldsX[rr * ADIM + col] * sg;
            }
        }
        // reduce the 16 column-lanes of each half (xor of bits 0..3 only)
        #pragma unroll
        for (int e = 0; e < 8; ++e) {
            #pragma unroll
            for (int off = 1; off < 16; off <<= 1)
                part[e] += __shfl_xor(part[e], off, 32);
        }
        if (m == 0) {
            #pragma unroll
            for (int e = 0; e < 8; ++e)
                s_out[row0 + hi * 8 + e] = part[e];
        }
    }
}

// ---------------------------------------------------------------------------
// Kernel 3: per-bag stable softmax + weighted pooling (bandwidth-bound).
// One workgroup per bag; bags are 1024 contiguous instances (CSR pointers).
// ---------------------------------------------------------------------------
__global__ __launch_bounds__(256)
void pool_kernel(const float* __restrict__ H, const long long* __restrict__ bag_ptr,
                 const float* __restrict__ s, float* __restrict__ Z) {
    const int bag = blockIdx.x;
    const int tid = threadIdx.x;
    const long long start = bag_ptr[bag];

    __shared__ float a[1024];
    __shared__ float red[256];

    const float* sb = s + start;
    v4f sv = *(const v4f*)(sb + tid * 4);
    float mx = fmaxf(fmaxf(sv[0], sv[1]), fmaxf(sv[2], sv[3]));
    red[tid] = mx;
    __syncthreads();
    for (int off = 128; off > 0; off >>= 1) {
        if (tid < off) red[tid] = fmaxf(red[tid], red[tid + off]);
        __syncthreads();
    }
    mx = red[0];
    __syncthreads();

    v4f ev;
    ev[0] = expf(sv[0] - mx); ev[1] = expf(sv[1] - mx);
    ev[2] = expf(sv[2] - mx); ev[3] = expf(sv[3] - mx);
    red[tid] = ev[0] + ev[1] + ev[2] + ev[3];
    __syncthreads();
    for (int off = 128; off > 0; off >>= 1) {
        if (tid < off) red[tid] += red[tid + off];
        __syncthreads();
    }
    const float inv = 1.0f / red[0];
    *(v4f*)(a + tid * 4) = ev * inv;
    __syncthreads();

    // thread t owns columns 4t..4t+3 -> fully coalesced 16B/lane rows
    const float* Hb = H + (size_t)start * DDIM + tid * 4;
    v4f acc = {};
    #pragma unroll 4
    for (int i = 0; i < 1024; ++i) {
        __builtin_prefetch(Hb + (size_t)(i + 24) * DDIM, 0, 0);
        const float w = a[i];
        v4f h = *(const v4f*)(Hb + (size_t)i * DDIM);
        acc += w * h;
    }
    *(v4f*)(Z + bag * DDIM + tid * 4) = acc;
}

// ---------------------------------------------------------------------------
extern "C" void kernel_launch(void* const* d_in, const int* in_sizes, int n_in,
                              void* d_out, int out_size, void* d_ws, size_t ws_size,
                              hipStream_t stream) {
    const float*     H       = (const float*)d_in[0];
    const long long* bag_ptr = (const long long*)d_in[1];
    const float*     Vw      = (const float*)d_in[2];
    const float*     Vb      = (const float*)d_in[3];
    const float*     Uw      = (const float*)d_in[4];
    const float*     Ub      = (const float*)d_in[5];
    const float*     ww      = (const float*)d_in[6];
    float*           Z       = (float*)d_out;

    char* ws = (char*)d_ws;
    __bf16* Vt = (__bf16*)ws;                              // 512 KB
    __bf16* Ut = (__bf16*)(ws + (size_t)ADIM * DDIM * 2);  // 512 KB
    float*  sL = (float*)(ws + (size_t)ADIM * DDIM * 4);   // 256 KB logits

    wcvt_kernel<<<(ADIM * DDIM + 255) / 256, 256, 0, stream>>>(Vw, Uw, Vt, Ut);
    logits_kernel<<<N_INST / ROWS_PER_WG, 256, 0, stream>>>(H, Vt, Ut, Vb, Ub, ww, sL);
    pool_kernel<<<NBAG, 256, 0, stream>>>(H, bag_ptr, sL, Z);
}
